// Policy_LSTM_41747082117874
// MI455X (gfx1250) — compile-verified, matrix operand order checked by reading, not run on hardware
//
#include <hip/hip_runtime.h>

#define H   64
#define G4  256   // 4*H
#define EDIM 2
#define TSTEPS 128
#define TILE_B 32
#define NTHREADS 256

typedef __attribute__((ext_vector_type(16))) __bf16 v16bf;
typedef __attribute__((ext_vector_type(8)))  __bf16 v8bf;
typedef __attribute__((ext_vector_type(8)))  float  v8f;

__device__ inline __bf16 f2bf(float f) {
    union { float f; unsigned u; } a; a.f = f;
    unsigned r = a.u + 0x7FFFu + ((a.u >> 16) & 1u);  // round-to-nearest-even
    union { unsigned short s; __bf16 b; } o; o.s = (unsigned short)(r >> 16);
    return o.b;
}
__device__ inline float sigf(float x) {
    return __builtin_amdgcn_rcpf(1.f + __expf(-x));
}
__device__ inline float tanh_fast(float x) {
    float e = __expf(-2.f * x);
    return (1.f - e) * __builtin_amdgcn_rcpf(1.f + e);
}

// 16x32 bf16 fragment from LDS (row-major, ld = H).
// ISA 7.12.2: lanes 0-15 hold K = kk+{0..7,16..23}, lanes 16-31 hold
// K = kk+{8..15,24..31}; row within tile = lane & 15.
__device__ inline v16bf ldfrag(const __bf16* base, int r0, int kk, int lane) {
    int r  = r0 + (lane & 15);
    int kb = kk + ((lane >> 4) << 3);
    const __bf16* p = base + r * H + kb;
    union { v16bf v; v8bf h[2]; } u;
    u.h[0] = *(const v8bf*)(p);
    u.h[1] = *(const v8bf*)(p + 16);
    return u.v;
}

// Same fragment layout, loaded straight from f32 weights in global memory
// with on-the-fly bf16 conversion (done once, before the t-loop).
__device__ inline v16bf ldfrag_g(const float* __restrict__ W, int r0, int kk, int lane) {
    int r  = r0 + (lane & 15);
    int kb = kk + ((lane >> 4) << 3);
    const float* p = W + r * H + kb;
    v16bf f;
    #pragma unroll
    for (int q = 0; q < 8; ++q) {
        f[q]     = f2bf(p[q]);
        f[q + 8] = f2bf(p[q + 16]);
    }
    return f;
}

__global__ __launch_bounds__(NTHREADS)
void lstm2_wmma_kernel(const float* __restrict__ x_all,
                       const float* __restrict__ Wih1, const float* __restrict__ Whh1,
                       const float* __restrict__ bih1, const float* __restrict__ bhh1,
                       const float* __restrict__ Wih2, const float* __restrict__ Whh2,
                       const float* __restrict__ bih2, const float* __restrict__ bhh2,
                       const float* __restrict__ Wlin, const float* __restrict__ blin,
                       float* __restrict__ out)
{
    __shared__ __bf16 sH1 [TILE_B * H];  // h1 state (bf16)
    __shared__ __bf16 sH2 [TILE_B * H];  // h2 state (bf16)
    __shared__ float  sG  [TILE_B * G4]; // gate pre-activations (f32)
    __shared__ float  sWx1[G4 * EDIM];   // W_ih1 (f32, E=2)
    __shared__ float  sB1 [G4];          // b_ih1 + b_hh1
    __shared__ float  sB2 [G4];          // b_ih2 + b_hh2
    __shared__ float  sWlin[H];
    __shared__ float  sX  [TILE_B * EDIM];

    const int tid  = threadIdx.x;
    const int lane = tid & 31;
    const int wave = tid >> 5;

    // wave -> 4 C tiles: BOTH M tiles x 2 consecutive N tiles.
    // Only 2 N tiles per wave -> 12 resident B-fragments (96 VGPRs),
    // keeping total allocation under 256 VGPRs (no s_set_vgpr_msb churn).
    const int nb = wave * 2;                  // base N tile (0..14)

    // ---- hoist this wave's weight B-fragments into registers ----
    v16bf bW1f[2][2], bWi2f[2][2], bWh2f[2][2];
    #pragma unroll
    for (int i = 0; i < 2; ++i) {
        #pragma unroll
        for (int k = 0; k < 2; ++k) {
            bW1f [i][k] = ldfrag_g(Whh1, (nb + i) * 16, k * 32, lane);
            bWi2f[i][k] = ldfrag_g(Wih2, (nb + i) * 16, k * 32, lane);
            bWh2f[i][k] = ldfrag_g(Whh2, (nb + i) * 16, k * 32, lane);
        }
    }

    // ---- small f32 staging ----
    for (int i = tid; i < G4 * EDIM; i += NTHREADS) sWx1[i] = Wih1[i];
    for (int i = tid; i < G4; i += NTHREADS) {
        sB1[i] = bih1[i] + bhh1[i];
        sB2[i] = bih2[i] + bhh2[i];
    }
    if (tid < H) sWlin[tid] = Wlin[tid];
    for (int i = tid; i < TILE_B * H; i += NTHREADS) {
        sH1[i] = f2bf(0.f);
        sH2[i] = f2bf(0.f);
    }

    float c1[8], c2[8], h2last[8];
    #pragma unroll
    for (int e = 0; e < 8; ++e) { c1[e] = 0.f; c2[e] = 0.f; h2last[e] = 0.f; }

    const int  row_e = tid >> 3;        // elementwise: batch row (0..31)
    const int  j0    = (tid & 7) * 8;   // 8 consecutive hidden units
    const long bbase = (long)blockIdx.x * TILE_B;

    const int col = (lane & 15);        // C-tile column (within tile)
    const int rhi = (lane >> 4) << 3;   // +8 rows for upper half-lanes

    __syncthreads();

    for (int t = 0; t < TSTEPS; ++t) {
        // ---- Phase A: stage x_t (+prefetch x_{t+1}), layer-1 WMMA -> sG ----
        if (tid < TILE_B) {
            long b = bbase + tid;
            const float* xp = x_all + (b * TSTEPS + t) * EDIM;
            sX[tid * 2 + 0] = xp[0];
            sX[tid * 2 + 1] = xp[1];
            __builtin_prefetch(xp + EDIM, 0, 3);   // next timestep's x
        }
        #pragma unroll
        for (int m = 0; m < 2; ++m) {
            v16bf a0 = ldfrag(sH1, m * 16, 0,  lane);
            v16bf a1 = ldfrag(sH1, m * 16, 32, lane);
            #pragma unroll
            for (int i = 0; i < 2; ++i) {
                v8f acc = {};
                acc = __builtin_amdgcn_wmma_f32_16x16x32_bf16(
                          false, a0, false, bW1f[i][0], (short)0, acc, false, false);
                acc = __builtin_amdgcn_wmma_f32_16x16x32_bf16(
                          false, a1, false, bW1f[i][1], (short)0, acc, false, false);
                int c0    = (nb + i) * 16 + col;
                int rbase = m * 16 + rhi;
                #pragma unroll
                for (int r = 0; r < 8; ++r)
                    sG[(rbase + r) * G4 + c0] = acc[r];
            }
        }
        __syncthreads();

        // ---- Phase B: layer-1 gates + LSTM update ----
        {
            float x0 = sX[row_e * 2 + 0], x1 = sX[row_e * 2 + 1];
            #pragma unroll
            for (int e = 0; e < 8; ++e) {
                int j = j0 + e;
                float gi = sG[row_e*G4 +       j] + x0*sWx1[(      j)*2] + x1*sWx1[(      j)*2+1] + sB1[      j];
                float gf = sG[row_e*G4 +  64 + j] + x0*sWx1[( 64 + j)*2] + x1*sWx1[( 64 + j)*2+1] + sB1[ 64 + j];
                float gg = sG[row_e*G4 + 128 + j] + x0*sWx1[(128 + j)*2] + x1*sWx1[(128 + j)*2+1] + sB1[128 + j];
                float go = sG[row_e*G4 + 192 + j] + x0*sWx1[(192 + j)*2] + x1*sWx1[(192 + j)*2+1] + sB1[192 + j];
                float c = sigf(gf) * c1[e] + sigf(gi) * tanh_fast(gg);
                c1[e] = c;
                float h = sigf(go) * tanh_fast(c);
                sH1[row_e * H + j] = f2bf(h);
            }
        }
        __syncthreads();

        // ---- Phase C: layer-2 WMMA  (h1@W_ih2^T + h2@W_hh2^T -> sG) ----
        #pragma unroll
        for (int m = 0; m < 2; ++m) {
            v16bf a10 = ldfrag(sH1, m * 16, 0,  lane);
            v16bf a11 = ldfrag(sH1, m * 16, 32, lane);
            v16bf a20 = ldfrag(sH2, m * 16, 0,  lane);
            v16bf a21 = ldfrag(sH2, m * 16, 32, lane);
            #pragma unroll
            for (int i = 0; i < 2; ++i) {
                v8f acc = {};
                acc = __builtin_amdgcn_wmma_f32_16x16x32_bf16(
                          false, a10, false, bWi2f[i][0], (short)0, acc, false, false);
                acc = __builtin_amdgcn_wmma_f32_16x16x32_bf16(
                          false, a11, false, bWi2f[i][1], (short)0, acc, false, false);
                acc = __builtin_amdgcn_wmma_f32_16x16x32_bf16(
                          false, a20, false, bWh2f[i][0], (short)0, acc, false, false);
                acc = __builtin_amdgcn_wmma_f32_16x16x32_bf16(
                          false, a21, false, bWh2f[i][1], (short)0, acc, false, false);
                int c0    = (nb + i) * 16 + col;
                int rbase = m * 16 + rhi;
                #pragma unroll
                for (int r = 0; r < 8; ++r)
                    sG[(rbase + r) * G4 + c0] = acc[r];
            }
        }
        __syncthreads();

        // ---- Phase D: layer-2 gates + LSTM update ----
        #pragma unroll
        for (int e = 0; e < 8; ++e) {
            int j = j0 + e;
            float gi = sG[row_e*G4 +       j] + sB2[      j];
            float gf = sG[row_e*G4 +  64 + j] + sB2[ 64 + j];
            float gg = sG[row_e*G4 + 128 + j] + sB2[128 + j];
            float go = sG[row_e*G4 + 192 + j] + sB2[192 + j];
            float c = sigf(gf) * c2[e] + sigf(gi) * tanh_fast(gg);
            c2[e] = c;
            float h = sigf(go) * tanh_fast(c);
            h2last[e] = h;
            sH2[row_e * H + j] = f2bf(h);
        }
        __syncthreads();
    }

    // ---- Final linear: out[b] = h2 . W_lin + b_lin ----
    float p = 0.f;
    #pragma unroll
    for (int e = 0; e < 8; ++e) p += h2last[e] * sWlin[j0 + e];
    sG[tid] = p;               // sG reused as reduction scratch
    __syncthreads();
    if (tid < TILE_B) {
        float s = blin[0];
        #pragma unroll
        for (int k = 0; k < 8; ++k) s += sG[tid * 8 + k];
        out[bbase + tid] = s;
    }
}

extern "C" void kernel_launch(void* const* d_in, const int* in_sizes, int n_in,
                              void* d_out, int out_size, void* d_ws, size_t ws_size,
                              hipStream_t stream) {
    const float* x_all = (const float*)d_in[0];
    const float* Wih1  = (const float*)d_in[1];
    const float* Whh1  = (const float*)d_in[2];
    const float* bih1  = (const float*)d_in[3];
    const float* bhh1  = (const float*)d_in[4];
    const float* Wih2  = (const float*)d_in[5];
    const float* Whh2  = (const float*)d_in[6];
    const float* bih2  = (const float*)d_in[7];
    const float* bhh2  = (const float*)d_in[8];
    const float* Wlin  = (const float*)d_in[9];
    const float* blin  = (const float*)d_in[10];
    float* out = (float*)d_out;

    int nB = in_sizes[0] / (TSTEPS * EDIM);       // 65536
    int grid = nB / TILE_B;                       // 2048 blocks, 8 waves each

    lstm2_wmma_kernel<<<grid, NTHREADS, 0, stream>>>(
        x_all, Wih1, Whh1, bih1, bhh1,
        Wih2, Whh2, bih2, bhh2, Wlin, blin, out);
}
